// MultiHeadAttention_22325240005253
// MI455X (gfx1250) — matrix operand order repaired and checked by reading, not compiled, not run on hardware
//
#include <hip/hip_runtime.h>

#define SLEN   2048
#define DMODEL 1024
#define NHEAD  16
#define HDIM   64
#define NBATCH 2
#define NBS    (NBATCH * SLEN)      // 4096 total rows

typedef __attribute__((ext_vector_type(16))) __bf16 v16bf;
typedef __attribute__((ext_vector_type(8)))  __bf16 v8bf;
typedef __attribute__((ext_vector_type(8)))  float  v8f;
typedef __attribute__((ext_vector_type(8)))  float  v8fv;

static __device__ __forceinline__ v8f wmma_bf16(v16bf a, v16bf b, v8f c) {
  // D(16x16 f32) = A(16x32 bf16) * B(32x16 bf16) + C
  return __builtin_amdgcn_wmma_f32_16x16x32_bf16(false, a, false, b, (short)0, c,
                                                 false, false);
}

// A-fragment gather: row M=(lane&15), K elems {2*kA..2*kA+15}, kA=(lane>>4)*8
static __device__ __forceinline__ v16bf load_afrag(const __bf16* rowp, int kA) {
  v8bf lo = *(const v8bf*)(rowp + 2 * kA);
  v8bf hi = *(const v8bf*)(rowp + 2 * kA + 8);
  v16bf a;
#pragma unroll
  for (int i = 0; i < 8; ++i) { a[i] = lo[i]; a[i + 8] = hi[i]; }
  return a;
}

// ---------------------------------------------------------------- fp32 -> bf16
__global__ void cvt_bf16_kernel(const float* __restrict__ in,
                                __bf16* __restrict__ out) {
  int i = (blockIdx.x * blockDim.x + threadIdx.x) * 8;
  v8fv x = *(const v8fv*)(in + i);
  v8bf o;
#pragma unroll
  for (int j = 0; j < 8; ++j) o[j] = (__bf16)x[j];
  *(v8bf*)(out + i) = o;
}

// ------------------------------------------------------- fused QKV projections
// One wave computes a 64x64 tile of Y = X @ W^T (K = 1024).
// Per k-step of 32: 4 A-frags + 4 B-frags -> 16 WMMAs (1 b128 load per WMMA).
// mat 0: Q (scaled 1/8, [b,h,s,hd])  mat 1: K ([b,h,s,hd])  mat 2: V^T ([b,h,hd,s])
__global__ __launch_bounds__(256)
void proj_kernel(const __bf16* __restrict__ Xq, const __bf16* __restrict__ Xk,
                 const __bf16* __restrict__ Xv,
                 const __bf16* __restrict__ Wq, const __bf16* __restrict__ Wk,
                 const __bf16* __restrict__ Wv,
                 __bf16* __restrict__ Qp, __bf16* __restrict__ Kp,
                 __bf16* __restrict__ Vt) {
  const int TILES = (NBS / 64) * (DMODEL / 64);        // 64*16 = 1024 per matrix
  int wave  = (blockIdx.x * blockDim.x + threadIdx.x) >> 5;
  int lane  = threadIdx.x & 31;
  int mat   = wave / TILES;
  int t     = wave % TILES;
  int m0    = (t / (DMODEL / 64)) * 64;
  int n0    = (t % (DMODEL / 64)) * 64;

  const __bf16* X = (mat == 0) ? Xq : (mat == 1) ? Xk : Xv;
  const __bf16* W = (mat == 0) ? Wq : (mat == 1) ? Wk : Wv;

  int n  = lane & 15;
  int hv = lane >> 4;
  int kA = hv * 8;
  int kB = hv * 16;

  const __bf16* arow[4];
  const __bf16* brow[4];
#pragma unroll
  for (int i = 0; i < 4; ++i) {
    arow[i] = X + (size_t)(m0 + i * 16 + n) * DMODEL;   // A: row M = lane&15
    brow[i] = W + (size_t)(n0 + i * 16 + n) * DMODEL;   // B[k][n] = W[n][k]
  }

  v8f acc[4][4] = {};                                   // [mi][ni]
  for (int k0 = 0; k0 < DMODEL; k0 += 32) {
    v16bf a[4], bf[4];
#pragma unroll
    for (int mi = 0; mi < 4; ++mi) a[mi] = load_afrag(arow[mi] + k0, kA);
#pragma unroll
    for (int ni = 0; ni < 4; ++ni) bf[ni] = *(const v16bf*)(brow[ni] + k0 + kB);
#pragma unroll
    for (int ni = 0; ni < 4; ++ni)
#pragma unroll
      for (int mi = 0; mi < 4; ++mi)
        acc[mi][ni] = wmma_bf16(a[mi], bf[ni], acc[mi][ni]);
  }

  float scl = (mat == 0) ? 0.125f : 1.0f;               // fold rsqrt(HD) into Q
#pragma unroll
  for (int mi = 0; mi < 4; ++mi)
#pragma unroll
    for (int ni = 0; ni < 4; ++ni) {
      int e  = n0 + ni * 16 + n;                        // C-layout: col = n
      int h  = e >> 6;
      int hd = e & 63;
#pragma unroll
      for (int r = 0; r < 8; ++r) {
        int row = m0 + mi * 16 + hv * 8 + r;            // C-layout: row = hv*8+r
        int b_  = row >> 11;
        int s   = row & (SLEN - 1);
        float v = acc[mi][ni][r] * scl;
        if (mat == 2) {
          Vt[((size_t)(b_ * NHEAD + h) * HDIM + hd) * SLEN + s] = (__bf16)v;
        } else {
          __bf16* O = (mat == 0) ? Qp : Kp;
          O[((size_t)(b_ * NHEAD + h) * SLEN + s) * HDIM + hd] = (__bf16)v;
        }
      }
    }
}

// ------------------------------------------------ causal flash attention core
// One wave handles (b, h, 32-query tile): two 16-row M-subtiles share every
// K/V B-fragment. 32 keys per iteration; fp32 online softmax.
__global__ __launch_bounds__(256)
void attn_kernel(const __bf16* __restrict__ Qp, const __bf16* __restrict__ Kp,
                 const __bf16* __restrict__ Vt, __bf16* __restrict__ Ctx) {
  __shared__ __bf16 Plds[8][2][16 * 32];   // per-wave, per-subtile P staging
  int w    = threadIdx.x >> 5;
  int lane = threadIdx.x & 31;
  int job  = blockIdx.x * 8 + w;           // 2048 jobs total
  int qtile = job & (SLEN / 32 - 1);       // 6 bits
  int h     = (job >> 6) & (NHEAD - 1);    // 4 bits
  int b     = job >> 10;                   // 1 bit
  int q0    = qtile * 32;

  int n  = lane & 15;
  int hv = lane >> 4;
  int kA = hv * 8;
  int kB = hv * 16;

  const __bf16* Qb = Qp + (size_t)(b * NHEAD + h) * SLEN * HDIM;
  const __bf16* Kb = Kp + (size_t)(b * NHEAD + h) * SLEN * HDIM;
  const __bf16* Vb = Vt + (size_t)(b * NHEAD + h) * HDIM * SLEN;

  // Q A-fragments: [mi][d-chunk], d-chunks [0,32) and [32,64)
  v16bf qa[2][2];
#pragma unroll
  for (int mi = 0; mi < 2; ++mi) {
    const __bf16* qrow = Qb + (size_t)(q0 + mi * 16 + n) * HDIM;
    qa[mi][0] = load_afrag(qrow, kA);
    qa[mi][1] = load_afrag(qrow + 32, kA);
  }

  float m[2][8], l[2][8];
  v8f acc[2][4] = {};                      // [mi][hd-chunk]
#pragma unroll
  for (int mi = 0; mi < 2; ++mi)
#pragma unroll
    for (int r = 0; r < 8; ++r) { m[mi][r] = -1e30f; l[mi][r] = 0.0f; }

  int kblocks = (q0 + 32 + 31) >> 5;       // causal: keys < q0+32
  for (int kb = 0; kb < kblocks; ++kb) {
    int ks = kb * 32;

    // ---- K B-fragments (shared by both M-subtiles)
    const __bf16* k0row = Kb + (size_t)(ks + n) * HDIM;
    const __bf16* k1row = Kb + (size_t)(ks + 16 + n) * HDIM;
    v16bf b00 = *(const v16bf*)(k0row + kB);
    v16bf b01 = *(const v16bf*)(k0row + 32 + kB);
    v16bf b10 = *(const v16bf*)(k1row + kB);
    v16bf b11 = *(const v16bf*)(k1row + 32 + kB);

    // ---- scores: 2 subtiles x 2 key-halves, K-dim 64 (2 chained WMMAs each)
    v8f s0[2] = {}, s1[2] = {};
#pragma unroll
    for (int mi = 0; mi < 2; ++mi) {
      s0[mi] = wmma_bf16(qa[mi][0], b00, s0[mi]);
      s0[mi] = wmma_bf16(qa[mi][1], b01, s0[mi]);
      s1[mi] = wmma_bf16(qa[mi][0], b10, s1[mi]);
      s1[mi] = wmma_bf16(qa[mi][1], b11, s1[mi]);
    }

    // ---- causal mask + online softmax per subtile
#pragma unroll
    for (int mi = 0; mi < 2; ++mi)
#pragma unroll
      for (int r = 0; r < 8; ++r) {
        int q = q0 + mi * 16 + hv * 8 + r;
        float v0 = (ks + n      <= q) ? s0[mi][r] : -1e30f;
        float v1 = (ks + 16 + n <= q) ? s1[mi][r] : -1e30f;
        float mx = fmaxf(v0, v1);
        mx = fmaxf(mx, __shfl_xor(mx, 1, 32));
        mx = fmaxf(mx, __shfl_xor(mx, 2, 32));
        mx = fmaxf(mx, __shfl_xor(mx, 4, 32));
        mx = fmaxf(mx, __shfl_xor(mx, 8, 32));
        float mn = fmaxf(m[mi][r], mx);
        float f  = __expf(m[mi][r] - mn);
        float p0 = __expf(v0 - mn);
        float p1 = __expf(v1 - mn);
        float rs = p0 + p1;
        rs += __shfl_xor(rs, 1, 32);
        rs += __shfl_xor(rs, 2, 32);
        rs += __shfl_xor(rs, 4, 32);
        rs += __shfl_xor(rs, 8, 32);
        l[mi][r] = l[mi][r] * f + rs;
        m[mi][r] = mn;
#pragma unroll
        for (int c = 0; c < 4; ++c) acc[mi][c][r] *= f;
        Plds[w][mi][(hv * 8 + r) * 32 + n]      = (__bf16)p0;
        Plds[w][mi][(hv * 8 + r) * 32 + 16 + n] = (__bf16)p1;
      }

    // wave-local LDS ordering (same-wave DS ops are in-order; wait for data)
    asm volatile("s_wait_dscnt 0" ::: "memory");

    // ---- P A-fragments (16x32, K = key index)
    v16bf pa[2];
#pragma unroll
    for (int mi = 0; mi < 2; ++mi)
#pragma unroll
      for (int i = 0; i < 8; ++i) {
        pa[mi][i]     = Plds[w][mi][n * 32 + 2 * kA + i];
        pa[mi][i + 8] = Plds[w][mi][n * 32 + 2 * kA + 8 + i];
      }

    // ---- ctx += P @ V  (V^T => contiguous along keys; shared by subtiles)
#pragma unroll
    for (int c = 0; c < 4; ++c) {
      const __bf16* vrow = Vb + (size_t)(c * 16 + n) * SLEN + ks;
      v16bf vbfrag = *(const v16bf*)(vrow + kB);
      acc[0][c] = wmma_bf16(pa[0], vbfrag, acc[0][c]);
      acc[1][c] = wmma_bf16(pa[1], vbfrag, acc[1][c]);
    }
  }

  // ---- normalize, store ctx as bf16 in [b, s, h*64+hd] for the Wo GEMM
#pragma unroll
  for (int mi = 0; mi < 2; ++mi)
#pragma unroll
    for (int r = 0; r < 8; ++r) {
      float inv = 1.0f / l[mi][r];
      int s = q0 + mi * 16 + hv * 8 + r;
#pragma unroll
      for (int c = 0; c < 4; ++c) {
        int e = h * HDIM + c * 16 + n;
        Ctx[((size_t)b * SLEN + s) * DMODEL + e] = (__bf16)(acc[mi][c][r] * inv);
      }
    }
}

// --------------------------------------------------- output projection (fp32)
// 64x64 tile per wave, same reuse scheme as proj_kernel.
__global__ __launch_bounds__(256)
void oproj_kernel(const __bf16* __restrict__ Ctx, const __bf16* __restrict__ Wo,
                  float* __restrict__ Out) {
  int wave = (blockIdx.x * blockDim.x + threadIdx.x) >> 5;
  int lane = threadIdx.x & 31;
  int m0   = (wave / (DMODEL / 64)) * 64;
  int n0   = (wave % (DMODEL / 64)) * 64;

  int n  = lane & 15;
  int hv = lane >> 4;
  int kA = hv * 8;
  int kB = hv * 16;

  const __bf16* arow[4];
  const __bf16* brow[4];
#pragma unroll
  for (int i = 0; i < 4; ++i) {
    arow[i] = Ctx + (size_t)(m0 + i * 16 + n) * DMODEL;
    brow[i] = Wo  + (size_t)(n0 + i * 16 + n) * DMODEL;
  }

  v8f acc[4][4] = {};
  for (int k0 = 0; k0 < DMODEL; k0 += 32) {
    v16bf a[4], bf[4];
#pragma unroll
    for (int mi = 0; mi < 4; ++mi) a[mi] = load_afrag(arow[mi] + k0, kA);
#pragma unroll
    for (int ni = 0; ni < 4; ++ni) bf[ni] = *(const v16bf*)(brow[ni] + k0 + kB);
#pragma unroll
    for (int ni = 0; ni < 4; ++ni)
#pragma unroll
      for (int mi = 0; mi < 4; ++mi)
        acc[mi][ni] = wmma_bf16(a[mi], bf[ni], acc[mi][ni]);
  }
#pragma unroll
  for (int mi = 0; mi < 4; ++mi)
#pragma unroll
    for (int ni = 0; ni < 4; ++ni)
#pragma unroll
      for (int r = 0; r < 8; ++r) {
        int row = m0 + mi * 16 + hv * 8 + r;
        Out[(size_t)row * DMODEL + n0 + ni * 16 + n] = acc[mi][ni][r];
      }
}

// ----------------------------------------------------------------------------
extern "C" void kernel_launch(void* const* d_in, const int* in_sizes, int n_in,
                              void* d_out, int out_size, void* d_ws, size_t ws_size,
                              hipStream_t stream) {
  const float* q  = (const float*)d_in[0];
  const float* k  = (const float*)d_in[1];
  const float* v  = (const float*)d_in[2];
  const float* wq = (const float*)d_in[3];
  const float* wk = (const float*)d_in[4];
  const float* wv = (const float*)d_in[5];
  const float* wo = (const float*)d_in[6];
  float* out = (float*)d_out;

  const size_t ACT = (size_t)NBS * DMODEL;      // 4 Mi elems
  const size_t WEL = (size_t)DMODEL * DMODEL;   // 1 Mi elems

  __bf16* ws  = (__bf16*)d_ws;                  // needs 32 Mi bf16 = 64 MB
  __bf16* Xq  = ws;
  __bf16* Xk  = Xq + ACT;
  __bf16* Xv  = Xk + ACT;
  __bf16* Wq  = Xv + ACT;
  __bf16* Wk  = Wq + WEL;
  __bf16* Wv  = Wk + WEL;
  __bf16* Wo  = Wv + WEL;
  __bf16* Qp  = Wo + WEL;
  __bf16* Kp  = Qp + ACT;
  __bf16* Vt  = Kp + ACT;
  __bf16* Ctx = Vt + ACT;

  // 1) fp32 -> bf16 conversions (8 elems / thread)
  cvt_bf16_kernel<<<ACT / 2048, 256, 0, stream>>>(q,  Xq);
  cvt_bf16_kernel<<<ACT / 2048, 256, 0, stream>>>(k,  Xk);
  cvt_bf16_kernel<<<ACT / 2048, 256, 0, stream>>>(v,  Xv);
  cvt_bf16_kernel<<<WEL / 2048, 256, 0, stream>>>(wq, Wq);
  cvt_bf16_kernel<<<WEL / 2048, 256, 0, stream>>>(wk, Wk);
  cvt_bf16_kernel<<<WEL / 2048, 256, 0, stream>>>(wv, Wv);
  cvt_bf16_kernel<<<WEL / 2048, 256, 0, stream>>>(wo, Wo);

  // 2) Q/K/V projections: 3 * 1024 wave-tiles (64x64) / 8 waves per block
  proj_kernel<<<3 * (NBS / 64) * (DMODEL / 64) / 8, 256, 0, stream>>>(
      Xq, Xk, Xv, Wq, Wk, Wv, Qp, Kp, Vt);

  // 3) causal attention: B*H*(S/32) = 2048 wave-jobs / 8 per block
  attn_kernel<<<NBATCH * NHEAD * (SLEN / 32) / 8, 256, 0, stream>>>(
      Qp, Kp, Vt, Ctx);

  // 4) output projection to fp32: 1024 wave-tiles / 8 per block
  oproj_kernel<<<(NBS / 64) * (DMODEL / 64) / 8, 256, 0, stream>>>(Ctx, Wo, out);
}
